// Attention_57294863728922
// MI455X (gfx1250) — matrix operand order repaired and checked by reading, not compiled, not run on hardware
//
#include <hip/hip_runtime.h>
#include <hip/hip_bf16.h>

typedef __attribute__((ext_vector_type(2)))  float  v2f;
typedef __attribute__((ext_vector_type(8)))  float  v8f;
typedef __attribute__((ext_vector_type(4)))  __bf16 v4bf;
typedef __attribute__((ext_vector_type(8)))  __bf16 v8bf;
typedef __attribute__((ext_vector_type(16))) __bf16 v16bf;

// fp32 exact path: D = A(16x4)*B(4x16) + C
__device__ __forceinline__ v8f wmma4(v2f a, v2f b, v8f c) {
    return __builtin_amdgcn_wmma_f32_16x16x4_f32(false, a, false, b, (short)0, c,
                                                 false, false);
}
// bf16 fast path: D = A(16x32)*B(32x16) + C, fp32 accumulate
__device__ __forceinline__ v8f wmma32(v16bf a, v16bf b, v8f c) {
    return __builtin_amdgcn_wmma_f32_16x16x32_bf16(false, a, false, b, (short)0, c,
                                                   false, false);
}
__device__ __forceinline__ v16bf cat8(v8bf a, v8bf b) {
    return __builtin_shufflevector(a, b, 0, 1, 2, 3, 4, 5, 6, 7,
                                   8, 9, 10, 11, 12, 13, 14, 15);
}
// Async global -> LDS (128b), per-lane addresses; tracked by ASYNCcnt.
__device__ __forceinline__ void async_g2l_b128(unsigned lds_off, const void* gptr) {
    asm volatile("global_load_async_to_lds_b128 %0, %1, off"
                 :: "v"(lds_off), "v"((unsigned long long)(uintptr_t)gptr)
                 : "memory");
}
__device__ __forceinline__ void wait_asynccnt0() {
    asm volatile("s_wait_asynccnt 0" ::: "memory");
}

#define N_EMBD 1024
#define SEQ    2048
#define NHEAD  16
#define HDIM   64

// ---------------------------------------------------------------------------
// GEMM [4096 x 1024] * [1024 x Nw] + bias, bf16x2 split precision:
//   x = hi + lo (bf16 each); A*B ~= Ah*Bh + Ah*Bl + Al*Bh  (fp32 accum)
// 3x V_WMMA_F32_16X16X32_BF16 per K=32 chunk per 16x16 tile (~1e-5 rel err).
// Block: 256 threads / 8 waves; tile 128(M) x 64(N); K chunk 32.
// MODE 0: QKV -> scatter q to qbuf[B,H,S,64], k/v to present[B,2,H,S,64].
// MODE 1: plain store + bias (projection).
// ---------------------------------------------------------------------------
template <int MODE>
__global__ __launch_bounds__(256)
void gemm_bf16x2_wmma(const float* __restrict__ A, const float* __restrict__ W,
                      const float* __restrict__ bias, float* __restrict__ out0,
                      float* __restrict__ out1, int Nw) {
    // stride 40 bf16 = 80B rows: every fragment load stays 16B-aligned.
    __shared__ __bf16 Ah[128 * 40];
    __shared__ __bf16 Al[128 * 40];
    __shared__ __bf16 Bh[64 * 40];   // transposed [N][K]
    __shared__ __bf16 Bl[64 * 40];

    const int tid  = threadIdx.x;
    const int wv   = tid >> 5;
    const int lane = tid & 31;
    const int half = lane >> 4;
    const int l16  = lane & 15;
    const int n0   = blockIdx.x * 64;
    const int m0   = blockIdx.y * 128;

    v8f acc[4] = {};

    for (int k0 = 0; k0 < N_EMBD; k0 += 32) {
        // Pull next K-chunk toward the WGP while this chunk computes.
        if (k0 + 32 < N_EMBD) {
            __builtin_prefetch(A + (size_t)(m0 + (tid >> 3)) * N_EMBD +
                                   (k0 + 32) + (tid & 7) * 4, 0, 1);
            __builtin_prefetch(W + (size_t)(k0 + 32 + (tid >> 4)) * Nw +
                                   n0 + (tid & 15) * 4, 0, 1);
        }
        __syncthreads();
        // Stage A tile 128x32: load fp32, split into hi/lo bf16.
        #pragma unroll
        for (int rr = 0; rr < 4; ++rr) {
            int idx = tid + 256 * rr;                 // 0..1023
            int row = idx >> 3, c4 = (idx & 7) * 4;
            float4 v = *(const float4*)(A + (size_t)(m0 + row) * N_EMBD + k0 + c4);
            v4bf hi, lo;
            hi.x = (__bf16)v.x; lo.x = (__bf16)(v.x - (float)hi.x);
            hi.y = (__bf16)v.y; lo.y = (__bf16)(v.y - (float)hi.y);
            hi.z = (__bf16)v.z; lo.z = (__bf16)(v.z - (float)hi.z);
            hi.w = (__bf16)v.w; lo.w = (__bf16)(v.w - (float)hi.w);
            *(v4bf*)(&Ah[row * 40 + c4]) = hi;
            *(v4bf*)(&Al[row * 40 + c4]) = lo;
        }
        // Stage B tile 32x64: load fp32 row-major, store transposed [N][K].
        #pragma unroll
        for (int rr = 0; rr < 2; ++rr) {
            int idx = tid + 256 * rr;                 // 0..511
            int row = idx >> 4, c4 = (idx & 15) * 4;  // row = K, c4 = N offset
            float4 v = *(const float4*)(W + (size_t)(k0 + row) * Nw + n0 + c4);
            float f[4] = {v.x, v.y, v.z, v.w};
            #pragma unroll
            for (int i = 0; i < 4; ++i) {
                __bf16 h = (__bf16)f[i];
                __bf16 l = (__bf16)(f[i] - (float)h);
                Bh[(c4 + i) * 40 + row] = h;
                Bl[(c4 + i) * 40 + row] = l;
            }
        }
        __syncthreads();

        // A-fragments (16-bit A layout: lane=row l16, K = {8h..8h+7, 16+8h..}).
        const __bf16* ap = &Ah[(wv * 16 + l16) * 40 + half * 8];
        const __bf16* lp = &Al[(wv * 16 + l16) * 40 + half * 8];
        v16bf ah = cat8(*(const v8bf*)ap, *(const v8bf*)(ap + 16));
        v16bf al = cat8(*(const v8bf*)lp, *(const v8bf*)(lp + 16));

        #pragma unroll
        for (int j = 0; j < 4; ++j) {
            // B-fragment (lane=col l16, K = {8h..8h+7, 16+8h..} in Bt).
            const __bf16* bp  = &Bh[(j * 16 + l16) * 40 + half * 8];
            const __bf16* blp = &Bl[(j * 16 + l16) * 40 + half * 8];
            v16bf bh = cat8(*(const v8bf*)bp,  *(const v8bf*)(bp + 16));
            v16bf bl = cat8(*(const v8bf*)blp, *(const v8bf*)(blp + 16));
            acc[j] = wmma32(ah, bh, acc[j]);
            acc[j] = wmma32(ah, bl, acc[j]);
            acc[j] = wmma32(al, bh, acc[j]);
        }
    }

    // Store (C layout: VGPR r holds M = r + 8*half, N = l16)
    #pragma unroll
    for (int j = 0; j < 4; ++j) {
        int n = n0 + j * 16 + l16;
        float bv = bias[n];
        #pragma unroll
        for (int r = 0; r < 8; ++r) {
            int m = m0 + wv * 16 + r + 8 * half;
            float val = acc[j][r] + bv;
            if (MODE == 1) {
                out0[(size_t)m * N_EMBD + n] = val;
            } else {
                int b = m >> 11, s = m & (SEQ - 1);
                int part = n >> 10, e = n & (N_EMBD - 1);
                int hd = e >> 6, d = e & 63;
                if (part == 0) {
                    out0[(((size_t)(b * NHEAD + hd)) * SEQ + s) * HDIM + d] = val;
                } else {
                    out1[((((size_t)(b * 2 + (part - 1))) * NHEAD + hd) * SEQ + s) * HDIM + d] = val;
                }
            }
        }
    }
}

// ---------------------------------------------------------------------------
// Sparse strided attention, flash-style online softmax (exact fp32 WMMA path).
// Block = (qb, head, batch): 128 queries x <=248 sparse keys.
// K/V tiles are gathered via async global->LDS (ASYNCcnt) per key index.
// ---------------------------------------------------------------------------
__global__ __launch_bounds__(256)
void sparse_attn_wmma(const float* __restrict__ qbuf,
                      const float* __restrict__ present,
                      float* __restrict__ hbuf) {
    __shared__ int   keyIdx[256];
    __shared__ float Kt[16 * 68];
    __shared__ float Vt[16 * 68];
    __shared__ float pbuf[8 * 16 * 17];

    const int tid  = threadIdx.x;
    const int wv   = tid >> 5;
    const int lane = tid & 31;
    const int half = lane >> 4;
    const int l16  = lane & 15;
    const int qb   = blockIdx.x;
    const int hd   = blockIdx.y;
    const int b    = blockIdx.z;

    const int nk = 8 * qb + 128;
    const int nt = (nk + 15) >> 4;

    {   // sparse key list: 8 summary cols per earlier block + local block
        int j = tid, v;
        if (j < 8 * qb)      v = (j >> 3) * 128 + 120 + (j & 7);
        else if (j < nk)     v = qb * 128 + (j - 8 * qb);
        else                 v = -1;
        keyIdx[j] = v;
    }

    const float* qrow =
        qbuf + (((size_t)(b * NHEAD + hd)) * SEQ + qb * 128 + wv * 16 + l16) * HDIM;
    v2f qa[16];
    #pragma unroll
    for (int ks = 0; ks < 16; ++ks) {
        qa[ks].x = qrow[ks * 4 + 2 * half];
        qa[ks].y = qrow[ks * 4 + 2 * half + 1];
    }

    const float* Kbase = present + (((size_t)(b * 2 + 0)) * NHEAD + hd) * SEQ * HDIM;
    const float* Vbase = present + (((size_t)(b * 2 + 1)) * NHEAD + hd) * SEQ * HDIM;

    float mrow[8], ssum[8];
    #pragma unroll
    for (int r = 0; r < 8; ++r) { mrow[r] = -3.0e38f; ssum[r] = 0.0f; }
    v8f o[4] = {};

    const int qrow0 = qb * 128 + wv * 16 + 8 * half;
    float* pw = &pbuf[wv * 16 * 17];

    for (int t = 0; t < nt; ++t) {
        __syncthreads();
        {   // async gather K/V tile (16 keys x 64). Padded keys clamp to 0:
            // their columns are masked to -3e38 below, so p = 0 kills them.
            int i = tid >> 4, c = (tid & 15) * 4;
            int key  = keyIdx[t * 16 + i];
            int keyc = (key < 0) ? 0 : key;
            unsigned kt_off = (unsigned)(uintptr_t)(&Kt[i * 68 + c]);
            unsigned vt_off = (unsigned)(uintptr_t)(&Vt[i * 68 + c]);
            async_g2l_b128(kt_off, Kbase + (size_t)keyc * HDIM + c);
            async_g2l_b128(vt_off, Vbase + (size_t)keyc * HDIM + c);
            wait_asynccnt0();
        }
        __syncthreads();

        // S = Q K^T over D=64 (16 fp32 WMMAs)
        v8f s = {};
        #pragma unroll
        for (int ks = 0; ks < 16; ++ks) {
            v2f bf;
            bf.x = Kt[l16 * 68 + ks * 4 + 2 * half];
            bf.y = Kt[l16 * 68 + ks * 4 + 2 * half + 1];
            s = wmma4(qa[ks], bf, s);
        }

        // scale + causal/pad mask + online softmax update
        int key = keyIdx[t * 16 + l16];
        #pragma unroll
        for (int r = 0; r < 8; ++r) {
            float sv = s[r] * 0.125f;
            int qg = qrow0 + r;
            if (key < 0 || key > qg) sv = -3.0e38f;
            float tm = sv;
            for (int msk = 1; msk < 16; msk <<= 1)
                tm = fmaxf(tm, __shfl_xor(tm, msk, 32));
            float mnew  = fmaxf(mrow[r], tm);
            float scale = __expf(mrow[r] - mnew);
            mrow[r] = mnew;
            ssum[r] *= scale;
            #pragma unroll
            for (int dc = 0; dc < 4; ++dc) o[dc][r] *= scale;
            float p = __expf(sv - mnew);
            ssum[r] += p;
            pw[(r + 8 * half) * 17 + l16] = p;
        }
        __syncthreads();

        // O += P V (16 fp32 WMMAs); P transposed via per-wave LDS buffer
        #pragma unroll
        for (int kc = 0; kc < 4; ++kc) {
            v2f af;
            af.x = pw[l16 * 17 + kc * 4 + 2 * half];
            af.y = pw[l16 * 17 + kc * 4 + 2 * half + 1];
            #pragma unroll
            for (int dc = 0; dc < 4; ++dc) {
                v2f bf;
                bf.x = Vt[(kc * 4 + 2 * half + 0) * 68 + dc * 16 + l16];
                bf.y = Vt[(kc * 4 + 2 * half + 1) * 68 + dc * 16 + l16];
                o[dc] = wmma4(af, bf, o[dc]);
            }
        }
    }

    #pragma unroll
    for (int r = 0; r < 8; ++r) {
        float tot = ssum[r];
        for (int msk = 1; msk < 16; msk <<= 1) tot += __shfl_xor(tot, msk, 32);
        float inv = 1.0f / tot;
        int srow = qrow0 + r;
        #pragma unroll
        for (int dc = 0; dc < 4; ++dc) {
            hbuf[((size_t)(b * SEQ + srow)) * N_EMBD + hd * HDIM + dc * 16 + l16] =
                o[dc][r] * inv;
        }
    }
}

// ---------------------------------------------------------------------------
extern "C" void kernel_launch(void* const* d_in, const int* in_sizes, int n_in,
                              void* d_out, int out_size, void* d_ws, size_t ws_size,
                              hipStream_t stream) {
    (void)in_sizes; (void)n_in; (void)out_size; (void)ws_size;
    const float* inputs = (const float*)d_in[0];
    const float* w_attn = (const float*)d_in[1];
    const float* b_attn = (const float*)d_in[2];
    const float* w_proj = (const float*)d_in[3];
    const float* b_proj = (const float*)d_in[4];

    float* out     = (float*)d_out;
    float* h_out   = out;                                 // [2,2048,1024]
    float* present = out + (size_t)2 * SEQ * N_EMBD;      // [2,2,16,2048,64]

    float* qbuf = (float*)d_ws;                           // [2,16,2048,64]
    float* hbuf = qbuf + (size_t)2 * NHEAD * SEQ * HDIM;  // [2,2048,1024]

    dim3 g1(3 * N_EMBD / 64, (2 * SEQ) / 128, 1);
    gemm_bf16x2_wmma<0><<<g1, 256, 0, stream>>>(inputs, w_attn, b_attn,
                                                qbuf, present, 3 * N_EMBD);

    dim3 g2(SEQ / 128, NHEAD, 2);
    sparse_attn_wmma<<<g2, 256, 0, stream>>>(qbuf, present, hbuf);

    dim3 g3(N_EMBD / 64, (2 * SEQ) / 128, 1);
    gemm_bf16x2_wmma<1><<<g3, 256, 0, stream>>>(hbuf, w_proj, b_proj,
                                                h_out, nullptr, N_EMBD);
}
